// GRU_35201551958675
// MI455X (gfx1250) — matrix operand order repaired and checked by reading, not compiled, not run on hardware
//
#include <hip/hip_runtime.h>
#include <hip/hip_bf16.h>
#include <math.h>

// ---------------- problem constants ----------------
#define TT    512
#define BATCH 256
#define I0    128
#define HD    256
#define G3    768            // 3*H, gate order r,z,n (PyTorch)

typedef __bf16 bf16;
typedef bf16  v16bf __attribute__((ext_vector_type(16)));
typedef float v8f   __attribute__((ext_vector_type(8)));

union Frag { uint4 u[2]; v16bf v; };

// A-fragment (16x32 bf16, M x K) from a row-major bf16 tile.
// ISA 7.12.2: lanes 0-15 hold row M=lane, K = k0..k0+7 and k0+16..k0+23;
// lanes 16-31 hold row M=lane-16, K = k0+8..k0+15 and k0+24..k0+31.
__device__ __forceinline__ v16bf load_fragA_rm(const char* base, int rowBytes,
                                               int k0, int lane) {
    int row = lane & 15;
    int off = row * rowBytes + (k0 + ((lane >> 4) << 3)) * 2;
    Frag f;
    f.u[0] = *(const uint4*)(base + off);
    f.u[1] = *(const uint4*)(base + off + 32);
    return f.v;
}

// B-fragment (32x16 bf16, K x N) from prepacked weights:
// packed[((nt*KC + kc)*32 + lane)*16 + i] -> 32 contiguous bytes per lane.
__device__ __forceinline__ v16bf load_fragB(const bf16* __restrict__ pW,
                                            int chunk, int lane) {
    const char* p = (const char*)pW + (size_t)chunk * 1024 + lane * 32;
    Frag f;
    f.u[0] = *(const uint4*)(p);
    f.u[1] = *(const uint4*)(p + 16);
    return f.v;
}

__device__ __forceinline__ v8f wmma_bf16(v16bf a, v16bf b, v8f c) {
    return __builtin_amdgcn_wmma_f32_16x16x32_bf16(false, a, false, b,
                                                   (short)0, c, false, false);
}

// Bulk async copy global->LDS: contiguous block, 16B per thread per chunk.
// VDST = LDS byte address (wave-relative == low 32 bits of generic pointer),
// VADDR = 64-bit global address. Tracked by ASYNCcnt.
__device__ __forceinline__ void stage_async(const bf16* __restrict__ src,
                                            unsigned ldsDst, int bytesTotal,
                                            int tid) {
    for (int e = tid * 16; e < bytesTotal; e += 256 * 16) {
        unsigned    d = ldsDst + e;
        const char* g = (const char*)src + e;
        asm volatile("global_load_async_to_lds_b128 %0, %1, off"
                     :: "v"(d), "v"(g) : "memory");
    }
}

__device__ __forceinline__ void wait_async0() {
    asm volatile("s_wait_asynccnt 0x0" ::: "memory");
}

// ---------------- x fp32 -> bf16 one-shot conversion ------------------------
__global__ void cvt_bf16(const float* __restrict__ in, bf16* __restrict__ out,
                         int total) {
    int idx = blockIdx.x * 256 + threadIdx.x;
    if (idx < total) out[idx] = (bf16)in[idx];
}

// ---------------- weight prepack: W[N,K] fp32 -> WMMA B-fragment bf16 -------
__global__ void pack_w(const float* __restrict__ W, bf16* __restrict__ out,
                       int K, int total) {
    int idx = blockIdx.x * 256 + threadIdx.x;
    if (idx >= total) return;
    int i     = idx & 15;
    int lane  = (idx >> 4) & 31;
    int chunk = idx >> 9;
    int KC    = K >> 5;
    int kc    = chunk % KC;
    int nt    = chunk / KC;
    int n = nt * 16 + (lane & 15);
    int k = kc * 32 + (i & 7) + ((i >> 3) << 4) + ((lane >> 4) << 3);
    out[idx] = (bf16)W[(size_t)n * K + k];
}

// ---------------- GRU scan: one workgroup owns 16 batch rows for all T ------
// LDS: sGI[16][768]f32 | sGH[16][768]f32 | sHF[16][256]f32 | sHB[16][256]bf16
//      | sX[2][16][In]bf16 (double-buffered, filled by async DMA)
__global__ void __launch_bounds__(256)
gru_scan(const bf16* __restrict__ xin, int In, int KCI,
         const bf16* __restrict__ pWih, const bf16* __restrict__ pWhh,
         const float* __restrict__ b_ih, const float* __restrict__ b_hh,
         bf16* __restrict__ seq_out, bf16* __restrict__ last_out) {
    extern __shared__ char smem[];
    float* sGI = (float*)smem;                 // 16*768
    float* sGH = sGI + 16 * G3;                // 16*768
    float* sHF = sGH + 16 * G3;                // 16*256
    bf16*  sHB = (bf16*)(sHF + 16 * HD);       // 16*256
    bf16*  sX  = sHB + 16 * HD;                // 2 * 16*In

    const int tid   = threadIdx.x;
    const int lane  = tid & 31;
    const int wid   = tid >> 5;
    const int b0    = blockIdx.x * 16;
    const int xbuf  = 16 * In;                 // elements per x buffer
    const int xbytes = 32 * In;                // bytes per x buffer
    const unsigned sXa = (unsigned)(unsigned long long)(void*)sX;

    // pull packed weights toward L2/WGP$ (global_prefetch_b8)
    if (tid < 64) {
        __builtin_prefetch(pWih + (size_t)tid * 1024, 0, 1);
        __builtin_prefetch(pWhh + (size_t)tid * 2048, 0, 1);
    }

    for (int e = tid; e < 16 * HD; e += 256) { sHF[e] = 0.f; sHB[e] = (bf16)0.f; }

    // prologue: kick off async stage of x_0 into buffer 0
    stage_async(xin + (size_t)b0 * In, sXa, xbytes, tid);

    for (int t = 0; t < TT; ++t) {
        wait_async0();          // x_t staging (issued last iteration) complete
        __syncthreads();
        // overlap: start DMA of x_{t+1} into the other buffer now
        if (t + 1 < TT)
            stage_async(xin + (size_t)(t + 1) * (BATCH * In) + (size_t)b0 * In,
                        sXa + ((t + 1) & 1) * xbytes, xbytes, tid);
        const bf16* cur = sX + (t & 1) * xbuf;

        // each wave computes 6 N-tiles of the [16 x 768] gate pre-activations
        #pragma unroll
        for (int j = 0; j < 6; ++j) {
            int nt  = wid * 6 + j;
            int col = nt * 16 + (lane & 15);
            int mb  = (lane >> 4) << 3;

            v8f acc = {};
            for (int kc = 0; kc < KCI; ++kc) {          // gi = x_t @ W_ih^T
                v16bf a = load_fragA_rm((const char*)cur, In * 2, kc * 32, lane);
                v16bf b = load_fragB(pWih, nt * KCI + kc, lane);
                acc = wmma_bf16(a, b, acc);
            }
            #pragma unroll
            for (int v = 0; v < 8; ++v) sGI[(mb + v) * G3 + col] = acc[v];

            v8f acch = {};
            #pragma unroll
            for (int kc = 0; kc < 8; ++kc) {            // gh = h @ W_hh^T
                v16bf a = load_fragA_rm((const char*)sHB, HD * 2, kc * 32, lane);
                v16bf b = load_fragB(pWhh, nt * 8 + kc, lane);
                acch = wmma_bf16(a, b, acch);
            }
            #pragma unroll
            for (int v = 0; v < 8; ++v) sGH[(mb + v) * G3 + col] = acch[v];
        }
        __syncthreads();

        // fp32 gate math: r,z,n -> h_new  (16 rows x 256 cols over 256 thr)
        {
            int row = tid >> 4;
            int cb  = (tid & 15) * 16;
            bf16* seqp = seq_out ? seq_out + (size_t)t * (BATCH * HD)
                                            + (size_t)(b0 + row) * HD : nullptr;
            const float* gi = sGI + row * G3;
            const float* gh = sGH + row * G3;
            for (int cc = 0; cc < 16; ++cc) {
                int c = cb + cc;
                float sr = (gi[c]       + b_ih[c])       + (gh[c]       + b_hh[c]);
                float sz = (gi[c + 256] + b_ih[c + 256]) + (gh[c + 256] + b_hh[c + 256]);
                float in_ =  gi[c + 512] + b_ih[c + 512];
                float hn  =  gh[c + 512] + b_hh[c + 512];
                float r = 1.f / (1.f + __expf(-sr));
                float z = 1.f / (1.f + __expf(-sz));
                float n = tanhf(in_ + r * hn);
                float hnew = (1.f - z) * n + z * sHF[row * HD + c];
                sHF[row * HD + c] = hnew;
                sHB[row * HD + c] = (bf16)hnew;
                if (seqp) seqp[c] = (bf16)hnew;
            }
        }
        __syncthreads();
    }

    if (last_out) {
        for (int e = tid; e < 16 * HD; e += 256)
            last_out[(size_t)b0 * HD + e] = (bf16)sHF[e];
    }
}

// ---------------- final FC: out = h2_last @ W_fc^T + b_fc (WMMA) ------------
__global__ void __launch_bounds__(256)
fc_wmma(const bf16* __restrict__ h2, const bf16* __restrict__ pWfc,
        const float* __restrict__ bias, float* __restrict__ out) {
    int tid = threadIdx.x, lane = tid & 31, wid = tid >> 5;
    int tile = blockIdx.x * 8 + wid;          // 256 tiles: 16 x 16
    int mt = tile >> 4, nt = tile & 15;
    int m0 = mt * 16, n0 = nt * 16;

    v8f acc = {};
    #pragma unroll
    for (int kc = 0; kc < 8; ++kc) {
        v16bf a = load_fragA_rm((const char*)(h2 + (size_t)m0 * HD),
                                HD * 2, kc * 32, lane);
        v16bf b = load_fragB(pWfc, nt * 8 + kc, lane);
        acc = wmma_bf16(a, b, acc);
    }
    float bn = bias[n0 + (lane & 15)];
    int mb = (lane >> 4) << 3;
    #pragma unroll
    for (int v = 0; v < 8; ++v)
        out[(size_t)(m0 + mb + v) * HD + n0 + (lane & 15)] = acc[v] + bn;
}

// ---------------- host launcher ---------------------------------------------
extern "C" void kernel_launch(void* const* d_in, const int* in_sizes, int n_in,
                              void* d_out, int out_size, void* d_ws, size_t ws_size,
                              hipStream_t stream) {
    const float* x     = (const float*)d_in[0];
    const float* W_ih0 = (const float*)d_in[1];
    const float* W_hh0 = (const float*)d_in[2];
    const float* b_ih0 = (const float*)d_in[3];
    const float* b_hh0 = (const float*)d_in[4];
    const float* W_ih1 = (const float*)d_in[5];
    const float* W_hh1 = (const float*)d_in[6];
    const float* b_ih1 = (const float*)d_in[7];
    const float* b_hh1 = (const float*)d_in[8];
    const float* W_fc  = (const float*)d_in[9];
    const float* b_fc  = (const float*)d_in[10];

    char* ws = (char*)d_ws;
    bf16* pWih0 = (bf16*)(ws);                       // 768*128 bf16
    bf16* pWhh0 = (bf16*)(ws + 196608);              // 768*256
    bf16* pWih1 = (bf16*)(ws + 589824);              // 768*256
    bf16* pWhh1 = (bf16*)(ws + 983040);              // 768*256
    bf16* pWfc  = (bf16*)(ws + 1376256);             // 256*256
    bf16* h1seq = (bf16*)(ws + 1507328);             // T*B*H bf16 (67 MB)
    bf16* h2last= (bf16*)(ws + 68616192);            // B*H bf16
    bf16* xbf   = (bf16*)(ws + 68747264);            // T*B*I bf16 (33.5 MB)

    // 1) prepack weights + convert x to bf16
    pack_w<<<(G3 * I0 + 255) / 256, 256, 0, stream>>>(W_ih0, pWih0, I0, G3 * I0);
    pack_w<<<(G3 * HD + 255) / 256, 256, 0, stream>>>(W_hh0, pWhh0, HD, G3 * HD);
    pack_w<<<(G3 * HD + 255) / 256, 256, 0, stream>>>(W_ih1, pWih1, HD, G3 * HD);
    pack_w<<<(G3 * HD + 255) / 256, 256, 0, stream>>>(W_hh1, pWhh1, HD, G3 * HD);
    pack_w<<<(HD * HD + 255) / 256, 256, 0, stream>>>(W_fc,  pWfc,  HD, HD * HD);
    cvt_bf16<<<(TT * BATCH * I0 + 255) / 256, 256, 0, stream>>>(
        x, xbf, TT * BATCH * I0);

    const size_t ldsFixed = (size_t)(16 * G3 * 2 + 16 * HD) * 4 + 16 * HD * 2;

    // 2) layer-0 scan (I=128): 16 independent batch slices, async x staging
    gru_scan<<<BATCH / 16, 256, ldsFixed + 2 * 32 * I0, stream>>>(
        xbf, I0, I0 / 32, pWih0, pWhh0, b_ih0, b_hh0, h1seq, nullptr);

    // 3) layer-1 scan (I=256) over bf16 h1 sequence, keep only final h2
    gru_scan<<<BATCH / 16, 256, ldsFixed + 2 * 32 * HD, stream>>>(
        h1seq, HD, HD / 32, pWih1, pWhh1, b_ih1, b_hh1, nullptr, h2last);

    // 4) final FC on last timestep -> fp32 d_out [B,H]
    fc_wmma<<<(BATCH / 16) * (HD / 16) / 8, 256, 0, stream>>>(
        h2last, pWfc, b_fc, (float*)d_out);
}
